// COGA_49366354100335
// MI455X (gfx1250) — matrix operand order repaired and checked by reading, not compiled, not run on hardware
//
#include <hip/hip_runtime.h>
#include <math.h>

// ---------------------------------------------------------------------------
// MCAN dual-stream transformer layer for MI455X (gfx1250).
// GEMMs: v_wmma_f32_16x16x32_bf16 (bf16 in, fp32 accum), operands staged into
// LDS by the Tensor Data Mover (tensor_load_to_lds, TENSORcnt tracked),
// double-buffered so DMA overlaps WMMA. Softmax / LayerNorm in fp32.
// ---------------------------------------------------------------------------

#define DEV static __device__ __forceinline__

typedef __attribute__((ext_vector_type(16))) __bf16 bf16x16;
typedef __attribute__((ext_vector_type(8)))  float  f32x8;
typedef __attribute__((ext_vector_type(4)))  unsigned int u32x4;
typedef __attribute__((ext_vector_type(8)))  unsigned int u32x8;
typedef __attribute__((ext_vector_type(8)))  int          i32x8;
typedef __attribute__((ext_vector_type(4)))  int          i32x4;

struct Bytes32 { u32x4 lo; u32x4 hi; };

static constexpr int BB  = 8;     // batch
static constexpr int S   = 512;   // seq len (both streams)
static constexpr int Hd  = 1024;  // hidden
static constexpr int NH  = 16;    // heads
static constexpr int DH  = 64;    // head dim
static constexpr int FF  = 4096;  // ffn hidden
static constexpr int NTOK = BB * S;                      // 4096 tokens
static constexpr long long ACT = (long long)NTOK * Hd;   // 4 Mi elements

DEV unsigned short f32_to_bf16(float f) {
  unsigned int u = __builtin_bit_cast(unsigned int, f);
  u += 0x7fffu + ((u >> 16) & 1u);            // round-to-nearest-even
  return (unsigned short)(u >> 16);
}

DEV f32x8 wmma_bf16(bf16x16 a, bf16x16 b, f32x8 c) {
  return __builtin_amdgcn_wmma_f32_16x16x32_bf16(
      /*neg_a=*/false, a, /*neg_b=*/false, b,
      /*c_mod=*/(short)0, c, /*reuse_a=*/false, /*reuse_b=*/false);
}

// ---------------------------------------------------------------------------
// TDM: DMA one [rows x 32] bf16 tile (row stride = strideElems) into LDS.
// D# per CDNA5 ISA ch.8: group0 = {count|flags, lds_addr, gaddr_lo,
// gaddr_hi|type=2}; group1 = {data_size=2B, tensor_dim0/1, tile_dim0/1,
// tensor_dim0_stride}; groups 2/3 zero (2-D tile). cpol=0.
// ---------------------------------------------------------------------------
DEV void tdm_load_tile(const unsigned short* gptr, unsigned lds_byte_off,
                       int rows, int strideElems) {
  unsigned long long ga = (unsigned long long)gptr;
  u32x4 g0;
  g0[0] = 1u;                                            // count=1, user desc
  g0[1] = lds_byte_off;                                  // lds_addr (bytes)
  g0[2] = (unsigned)ga;                                  // global_addr[31:0]
  g0[3] = (unsigned)((ga >> 32) & 0x1FFFFFFu) | (2u << 30); // [56:32] | type=2

  u32x8 g1 = {};
  g1[0] = (1u << 16);                 // workgroup_mask=0, data_size=1 (2B)
  g1[1] = (32u << 16);                // tensor_dim0 = 32 (bits 79:48, lo16)
  g1[2] = ((unsigned)rows << 16);     // tensor_dim0 hi16=0 | tensor_dim1 lo16
  g1[3] = (32u << 16);                // tensor_dim1 hi16=0 | tile_dim0 = 32
  g1[4] = (unsigned)rows;             // tile_dim1 = rows, tile_dim2 = 0
  g1[5] = (unsigned)strideElems;      // tensor_dim0_stride[31:0]
  g1[6] = 0u;                         // stride hi16 | tensor_dim1_stride lo16
  g1[7] = 0u;

  i32x4 z4 = {};
#if __clang_major__ >= 23
  i32x8 z8 = {};
  __builtin_amdgcn_tensor_load_to_lds(g0, __builtin_bit_cast(i32x8, g1),
                                      z4, z4, z8, 0);
#else
  __builtin_amdgcn_tensor_load_to_lds(g0, __builtin_bit_cast(i32x8, g1),
                                      z4, z4, 0);
#endif
}

DEV unsigned lds_off(const void* p) {
  // generic pointer to LDS: low 32 bits are the LDS byte offset
  return (unsigned)(unsigned long long)p;
}

// ---------------------------------------------------------------------------
// NT GEMM: C[M,N] = A[M,K] * BT[N,K]^T (+bias, opt. relu), batched.
// A, BT bf16 K-major. Block tile BM x BN, BK=32, WR x WC waves, per-wave
// tile (BM/WR) x (BN/WC) as MT x NTC grid of 16x16 WMMAs.
// Staging: TDM double-buffer, wave 0 issues, s_wait_tensorcnt + barrier.
// Requires M%BM==0, N%BN==0, K%32==0 (true for all shapes here).
// ---------------------------------------------------------------------------
template <int BM, int BN, int WR, int WC, bool BIAS, bool RELU>
__global__ __launch_bounds__(WR * WC * 32)
void k_gemm_tdm(const unsigned short* __restrict__ A,
                const unsigned short* __restrict__ BT,
                const float* __restrict__ bias,
                float* __restrict__ C,
                int M, int N, int K,
                long long strideA, long long strideB, long long strideC) {
  constexpr int MT  = BM / WR / 16;
  constexpr int NTC = BN / WC / 16;
  __shared__ __align__(16) unsigned short As[2][BM][32];
  __shared__ __align__(16) unsigned short Bs[2][BN][32];

  const int batch = blockIdx.z;
  A  += (long long)batch * strideA;
  BT += (long long)batch * strideB;
  C  += (long long)batch * strideC;

  const int m0   = blockIdx.y * BM;
  const int n0   = blockIdx.x * BN;
  const int tid  = threadIdx.x;
  const int wave = tid >> 5;
  const int lane = tid & 31;
  const int wm = (wave / WC) * (BM / WR);
  const int wn = (wave % WC) * (BN / WC);

  f32x8 acc[MT][NTC] = {};

  const int ml = lane & 15;
  const int ak = (lane >> 4) * 8;    // A frag K base (ISA 16-bit A layout)
  const int bk = (lane >> 4) * 8;    // B frag K base (same K order as A)
  const int nsteps = K >> 5;

  if (wave == 0) {  // prologue: DMA first K-slab of both operands
    tdm_load_tile(&A [(long long)m0 * K], lds_off(&As[0][0][0]), BM, K);
    tdm_load_tile(&BT[(long long)n0 * K], lds_off(&Bs[0][0][0]), BN, K);
  }

  for (int step = 0; step < nsteps; ++step) {
    const int buf = step & 1;
    if (wave == 0) {
      if (step + 1 < nsteps) {   // prefetch next slab into other buffer
        const long long k0 = (long long)(step + 1) << 5;
        tdm_load_tile(&A [(long long)m0 * K + k0],
                      lds_off(&As[buf ^ 1][0][0]), BM, K);
        tdm_load_tile(&BT[(long long)n0 * K + k0],
                      lds_off(&Bs[buf ^ 1][0][0]), BN, K);
        __builtin_amdgcn_s_wait_tensorcnt(2);  // current slab complete
      } else {
        __builtin_amdgcn_s_wait_tensorcnt(0);
      }
    }
    __syncthreads();

    bf16x16 af[MT], bfr[NTC];
#pragma unroll
    for (int i = 0; i < MT; ++i) {
      Bytes32 t;
      t.lo = *(const u32x4*)(&As[buf][wm + i * 16 + ml][ak]);
      t.hi = *(const u32x4*)(&As[buf][wm + i * 16 + ml][ak + 16]);
      af[i] = __builtin_bit_cast(bf16x16, t);
    }
#pragma unroll
    for (int j = 0; j < NTC; ++j) {
      Bytes32 u;
      u.lo = *(const u32x4*)(&Bs[buf][wn + j * 16 + ml][bk]);
      u.hi = *(const u32x4*)(&Bs[buf][wn + j * 16 + ml][bk + 16]);
      bfr[j] = __builtin_bit_cast(bf16x16, u);
    }
#pragma unroll
    for (int i = 0; i < MT; ++i)
#pragma unroll
      for (int j = 0; j < NTC; ++j)
        acc[i][j] = wmma_bf16(af[i], bfr[j], acc[i][j]);
    __syncthreads();
  }

  // epilogue (ISA 32-bit C/D layout: lane n=lane&15, row = r + (lane>>4)*8)
#pragma unroll
  for (int i = 0; i < MT; ++i)
#pragma unroll
    for (int j = 0; j < NTC; ++j) {
      const int col = n0 + wn + j * 16 + (lane & 15);
      float bv = 0.0f;
      if constexpr (BIAS) bv = bias[col];
#pragma unroll
      for (int r = 0; r < 8; ++r) {
        const int row = m0 + wm + i * 16 + ((lane >> 4) * 8) + r;
        float v = acc[i][j][r] + bv;
        if constexpr (RELU) v = fmaxf(v, 0.0f);
        C[(long long)row * N + col] = v;
      }
    }
}

// ---------------------------------------------------------------------------
// Elementwise / layout kernels
// ---------------------------------------------------------------------------
__global__ void k_cvt_bf16(const float* __restrict__ in,
                           unsigned short* __restrict__ out, long long n) {
  long long i = (long long)blockIdx.x * 256 + threadIdx.x;
  if (i < n) out[i] = f32_to_bf16(in[i]);
}

// in[K][N] fp32  ->  out[N][K] bf16   (weights: [din,dout] -> K-major)
__global__ void k_transpose_cvt(const float* __restrict__ in,
                                unsigned short* __restrict__ out,
                                int K, int N) {
  long long idx = (long long)blockIdx.x * 256 + threadIdx.x;
  if (idx >= (long long)K * N) return;
  int k = (int)(idx / N);
  int n = (int)(idx - (long long)k * N);
  out[(long long)n * K + k] = f32_to_bf16(in[idx]);
}

// qproj+rproj [NT,H] fp32 -> qrh bf16 [b,h,s,d], scaled by 1/sqrt(DH)
__global__ void k_split_qr(const float* __restrict__ q,
                           const float* __restrict__ r,
                           unsigned short* __restrict__ out) {
  long long idx = (long long)blockIdx.x * 256 + threadIdx.x;
  if (idx >= ACT) return;
  int row = (int)(idx >> 10), col = (int)(idx & 1023);
  int b = row >> 9, s = row & 511, h = col >> 6, d = col & 63;
  out[((((long long)(b * NH + h)) * S + s) << 6) + d] =
      f32_to_bf16((q[idx] + r[idx]) * 0.125f);
}

// kproj [NT,H] fp32 -> kh bf16 [b,h,s,d]
__global__ void k_split_k(const float* __restrict__ in,
                          unsigned short* __restrict__ out) {
  long long idx = (long long)blockIdx.x * 256 + threadIdx.x;
  if (idx >= ACT) return;
  int row = (int)(idx >> 10), col = (int)(idx & 1023);
  int b = row >> 9, s = row & 511, h = col >> 6, d = col & 63;
  out[((((long long)(b * NH + h)) * S + s) << 6) + d] = f32_to_bf16(in[idx]);
}

// vproj [NT,H] fp32 -> vht bf16 [b,h,d,s]  (K-major for att@V NT gemm)
__global__ void k_split_vt(const float* __restrict__ in,
                           unsigned short* __restrict__ out) {
  long long idx = (long long)blockIdx.x * 256 + threadIdx.x;
  if (idx >= ACT) return;
  int row = (int)(idx >> 10), col = (int)(idx & 1023);
  int b = row >> 9, s = row & 511, h = col >> 6, d = col & 63;
  out[(((long long)(b * NH + h)) * DH + d) * S + s] = f32_to_bf16(in[idx]);
}

// attout fp32 [b,h,s,d] -> merged bf16 [b*s, h*DH+d]
__global__ void k_merge(const float* __restrict__ in,
                        unsigned short* __restrict__ out) {
  long long idx = (long long)blockIdx.x * 256 + threadIdx.x;
  if (idx >= ACT) return;
  int row = (int)(idx >> 10), col = (int)(idx & 1023);
  int b = row >> 9, s = row & 511, h = col >> 6, d = col & 63;
  long long ini = ((((long long)(b * NH + h)) * S + s) << 6) + d;
  out[idx] = f32_to_bf16(in[ini]);
}

// masked softmax over 512 keys; block = 256 threads, one row per block.
__global__ __launch_bounds__(256)
void k_softmax_mask(const float* __restrict__ sc,
                    const unsigned char* __restrict__ mask,
                    unsigned short* __restrict__ att,
                    int rows_per_batch) {
  __shared__ float red[256];
  const int row = blockIdx.x;
  const int b = row / rows_per_batch;
  const float* sr = sc + (long long)row * 512;
  const unsigned char* mr = mask + (long long)b * 512;
  const int t = threadIdx.x;

  float v0 = mr[t]       ? -1e9f : sr[t];
  float v1 = mr[t + 256] ? -1e9f : sr[t + 256];

  red[t] = fmaxf(v0, v1);
  __syncthreads();
  for (int o = 128; o > 0; o >>= 1) {
    if (t < o) red[t] = fmaxf(red[t], red[t + o]);
    __syncthreads();
  }
  const float mx = red[0];
  __syncthreads();

  float e0 = __expf(v0 - mx), e1 = __expf(v1 - mx);
  red[t] = e0 + e1;
  __syncthreads();
  for (int o = 128; o > 0; o >>= 1) {
    if (t < o) red[t] += red[t + o];
    __syncthreads();
  }
  const float inv = 1.0f / red[0];
  att[(long long)row * 512 + t]       = f32_to_bf16(e0 * inv);
  att[(long long)row * 512 + t + 256] = f32_to_bf16(e1 * inv);
}

// out = g * (x+h - mean) / (std_unbiased + eps) + beta ; optional bf16 copy.
__global__ __launch_bounds__(256)
void k_add_ln(const float* __restrict__ x, const float* __restrict__ h,
              const float* __restrict__ g, const float* __restrict__ beta,
              float* __restrict__ out, unsigned short* __restrict__ outb) {
  __shared__ float red[256];
  const int row = blockIdx.x;
  const float* xr = x + (long long)row * Hd;
  const float* hr = h + (long long)row * Hd;
  const int t = threadIdx.x;

  float v[4]; float s = 0.0f;
#pragma unroll
  for (int i = 0; i < 4; ++i) {
    const int c = t + i * 256;
    v[i] = xr[c] + hr[c];
    s += v[i];
  }
  red[t] = s;
  __syncthreads();
  for (int o = 128; o > 0; o >>= 1) {
    if (t < o) red[t] += red[t + o];
    __syncthreads();
  }
  const float mean = red[0] * (1.0f / (float)Hd);
  __syncthreads();

  float ss = 0.0f;
#pragma unroll
  for (int i = 0; i < 4; ++i) { v[i] -= mean; ss += v[i] * v[i]; }
  red[t] = ss;
  __syncthreads();
  for (int o = 128; o > 0; o >>= 1) {
    if (t < o) red[t] += red[t + o];
    __syncthreads();
  }
  const float stdv = sqrtf(red[0] * (1.0f / (float)(Hd - 1)));
  const float inv = 1.0f / (stdv + 1e-6f);

#pragma unroll
  for (int i = 0; i < 4; ++i) {
    const int c = t + i * 256;
    const float o = g[c] * v[i] * inv + beta[c];
    out[(long long)row * Hd + c] = o;
    if (outb) outb[(long long)row * Hd + c] = f32_to_bf16(o);
  }
}

// ---------------------------------------------------------------------------
// Host orchestration
// ---------------------------------------------------------------------------
static void gemm_launch(const unsigned short* A, const unsigned short* BT,
                        const float* bias, float* C, int M, int N, int K,
                        int batch, long long sA, long long sB, long long sC,
                        bool relu, hipStream_t stream) {
  const bool big = (M % 128 == 0) && (N % 128 == 0);
  if (big) {
    dim3 g(N / 128, M / 128, batch), b(256);
    if (bias && relu)
      k_gemm_tdm<128,128,2,4,true, true ><<<g,b,0,stream>>>(A,BT,bias,C,M,N,K,sA,sB,sC);
    else if (bias)
      k_gemm_tdm<128,128,2,4,true, false><<<g,b,0,stream>>>(A,BT,bias,C,M,N,K,sA,sB,sC);
    else
      k_gemm_tdm<128,128,2,4,false,false><<<g,b,0,stream>>>(A,BT,bias,C,M,N,K,sA,sB,sC);
  } else {
    dim3 g(N / 64, M / 64, batch), b(128);
    if (bias && relu)
      k_gemm_tdm<64,64,2,2,true, true ><<<g,b,0,stream>>>(A,BT,bias,C,M,N,K,sA,sB,sC);
    else if (bias)
      k_gemm_tdm<64,64,2,2,true, false><<<g,b,0,stream>>>(A,BT,bias,C,M,N,K,sA,sB,sC);
    else
      k_gemm_tdm<64,64,2,2,false,false><<<g,b,0,stream>>>(A,BT,bias,C,M,N,K,sA,sB,sC);
  }
}

extern "C" void kernel_launch(void* const* d_in, const int* in_sizes, int n_in,
                              void* d_out, int out_size, void* d_ws, size_t ws_size,
                              hipStream_t stream) {
  (void)in_sizes; (void)n_in; (void)out_size; (void)ws_size;

  const float* P[64];
  for (int i = 0; i < 56; ++i) P[i] = (const float*)d_in[i];

  // setup_inputs() dict order:
  // 0:x 1:y 2:x_mask 3:y_mask 4:x_pos 5:y_pos, then params (insertion order):
  //   x2y(6..15), y2x(16..25), xself(26..35) each [v.w,v.b,k.w,k.b,q.w,q.b,r.w,r.b,m.w,m.b]
  //   ffnx(36..39)=[fc1.w,fc1.b,fc2.w,fc2.b], ffny(40..43), ln: 44+2i=g_i, 45+2i=b_i
  const float* x    = P[0];
  const float* y    = P[1];
  const unsigned char* xmask = (const unsigned char*)d_in[2];
  const unsigned char* ymask = (const unsigned char*)d_in[3];
  const float* xpos = P[4];
  const float* ypos = P[5];

  // ---- bump allocator over workspace ----
  char* wsp = (char*)d_ws;
  auto alloc = [&](size_t bytes) -> void* {
    void* p = (void*)wsp;
    wsp += (bytes + 255) & ~(size_t)255;
    return p;
  };

  unsigned short* mhaWT[3][5];
  const float*    mhaB [3][5];
  for (int mi = 0; mi < 3; ++mi)
    for (int li = 0; li < 5; ++li)
      mhaWT[mi][li] = (unsigned short*)alloc((size_t)Hd * Hd * 2);
  unsigned short* ffnWT[2][2];
  const float*    ffnB [2][2];
  for (int fi = 0; fi < 2; ++fi)
    for (int li = 0; li < 2; ++li)
      ffnWT[fi][li] = (unsigned short*)alloc((size_t)Hd * FF * 2);

  unsigned short* xb  = (unsigned short*)alloc(ACT * 2);
  unsigned short* yb  = (unsigned short*)alloc(ACT * 2);
  unsigned short* xpb = (unsigned short*)alloc(ACT * 2);
  unsigned short* ypb = (unsigned short*)alloc(ACT * 2);

  float* nx1 = (float*)alloc(ACT * 4);
  float* ny1 = (float*)alloc(ACT * 4);
  float* nx2 = (float*)alloc(ACT * 4);
  float* ny2 = (float*)alloc(ACT * 4);
  unsigned short* nx1b = (unsigned short*)alloc(ACT * 2);
  unsigned short* ny1b = (unsigned short*)alloc(ACT * 2);
  unsigned short* nx2b = (unsigned short*)alloc(ACT * 2);
  unsigned short* ny2b = (unsigned short*)alloc(ACT * 2);

  float* vproj = (float*)alloc(ACT * 4);
  float* kproj = (float*)alloc(ACT * 4);
  float* qproj = (float*)alloc(ACT * 4);
  float* rproj = (float*)alloc(ACT * 4);
  unsigned short* qrh = (unsigned short*)alloc(ACT * 2);
  unsigned short* kh  = (unsigned short*)alloc(ACT * 2);
  unsigned short* vht = (unsigned short*)alloc(ACT * 2);
  const long long SCORES = (long long)BB * NH * S * S;         // 32 Mi
  float*          scores = (float*)alloc(SCORES * 4);          // reused by ffn
  unsigned short* att    = (unsigned short*)alloc(SCORES * 2); // reused by ffn
  float*          attout = (float*)alloc(ACT * 4);
  unsigned short* merged = (unsigned short*)alloc(ACT * 2);
  float*          atted  = (float*)alloc(ACT * 4);

  float*          ffh  = scores;   // [NT, FF] fp32
  unsigned short* ffhb = att;      // [NT, FF] bf16

  const float* lnG[6]; const float* lnB[6];
  for (int i = 0; i < 6; ++i) { lnG[i] = P[44 + 2 * i]; lnB[i] = P[45 + 2 * i]; }

  // ---- weight transpose+convert ----
  for (int mi = 0; mi < 3; ++mi) {
    const int base = 6 + mi * 10;
    for (int li = 0; li < 5; ++li) {
      mhaB[mi][li] = P[base + li * 2 + 1];
      k_transpose_cvt<<<(Hd * Hd) / 256, 256, 0, stream>>>(
          P[base + li * 2], mhaWT[mi][li], Hd, Hd);
    }
  }
  for (int fi = 0; fi < 2; ++fi) {
    const int base = 36 + fi * 4;
    ffnB[fi][0] = P[base + 1];
    ffnB[fi][1] = P[base + 3];
    k_transpose_cvt<<<(Hd * FF) / 256, 256, 0, stream>>>(P[base + 0], ffnWT[fi][0], Hd, FF);
    k_transpose_cvt<<<(FF * Hd) / 256, 256, 0, stream>>>(P[base + 2], ffnWT[fi][1], FF, Hd);
  }

  // ---- activation conversions ----
  const int EB = (int)(ACT / 256);
  k_cvt_bf16<<<EB, 256, 0, stream>>>(x,    xb,  ACT);
  k_cvt_bf16<<<EB, 256, 0, stream>>>(y,    yb,  ACT);
  k_cvt_bf16<<<EB, 256, 0, stream>>>(xpos, xpb, ACT);
  k_cvt_bf16<<<EB, 256, 0, stream>>>(ypos, ypb, ACT);

  auto gemm = [&](const unsigned short* A, const unsigned short* BT,
                  const float* bias, float* C, int M, int N, int K, int batch,
                  long long sA, long long sB, long long sC, bool relu) {
    gemm_launch(A, BT, bias, C, M, N, K, batch, sA, sB, sC, relu, stream);
  };

  auto attn = [&](const unsigned short* qsrc, const unsigned short* kvsrc,
                  const unsigned short* rsrc, const unsigned char* mask,
                  unsigned short* const* wt, const float* const* bs,
                  float* out_atted) {
    gemm(kvsrc, wt[0], bs[0], vproj, NTOK, Hd, Hd, 1, 0, 0, 0, false);
    gemm(kvsrc, wt[1], bs[1], kproj, NTOK, Hd, Hd, 1, 0, 0, 0, false);
    gemm(qsrc,  wt[2], bs[2], qproj, NTOK, Hd, Hd, 1, 0, 0, 0, false);
    gemm(rsrc,  wt[3], bs[3], rproj, NTOK, Hd, Hd, 1, 0, 0, 0, false);
    k_split_qr<<<EB, 256, 0, stream>>>(qproj, rproj, qrh);
    k_split_k <<<EB, 256, 0, stream>>>(kproj, kh);
    k_split_vt<<<EB, 256, 0, stream>>>(vproj, vht);
    // scores = ((q+r)/sqrt(Dh)) @ k^T, batched over 128 (b,h)
    gemm(qrh, kh, nullptr, scores, S, S, DH, BB * NH,
         (long long)S * DH, (long long)S * DH, (long long)S * S, false);
    k_softmax_mask<<<BB * NH * S, 256, 0, stream>>>(scores, mask, att, NH * S);
    // out = att @ V   (V stored [Dh,Sk] K-major) -> 64x64-tile kernel (N=64)
    gemm(att, vht, nullptr, attout, S, DH, S, BB * NH,
         (long long)S * S, (long long)DH * S, (long long)S * DH, false);
    k_merge<<<EB, 256, 0, stream>>>(attout, merged);
    gemm(merged, wt[4], bs[4], out_atted, NTOK, Hd, Hd, 1, 0, 0, 0, false);
  };

  auto ffn = [&](const unsigned short* inb, int fi, float* outf) {
    gemm(inb, ffnWT[fi][0], ffnB[fi][0], ffh, NTOK, FF, Hd, 1, 0, 0, 0, true);
    k_cvt_bf16<<<(int)(((long long)NTOK * FF) / 256), 256, 0, stream>>>(
        ffh, ffhb, (long long)NTOK * FF);
    gemm(ffhb, ffnWT[fi][1], ffnB[fi][1], outf, NTOK, Hd, FF, 1, 0, 0, 0, false);
  };

  float* outx = (float*)d_out;
  float* outy = (float*)d_out + ACT;

  // cross attention
  attn(xb, yb, xpb, ymask, mhaWT[0], mhaB[0], atted);           // x2y
  k_add_ln<<<NTOK, 256, 0, stream>>>(x, atted, lnG[0], lnB[0], nx1, nx1b);
  attn(yb, xb, ypb, xmask, mhaWT[1], mhaB[1], atted);           // y2x
  k_add_ln<<<NTOK, 256, 0, stream>>>(y, atted, lnG[1], lnB[1], ny1, ny1b);
  // self attention (xself weights reused for both streams, per reference)
  attn(nx1b, nx1b, xpb, xmask, mhaWT[2], mhaB[2], atted);
  k_add_ln<<<NTOK, 256, 0, stream>>>(nx1, atted, lnG[2], lnB[2], nx2, nx2b);
  attn(ny1b, ny1b, ypb, ymask, mhaWT[2], mhaB[2], atted);
  k_add_ln<<<NTOK, 256, 0, stream>>>(ny1, atted, lnG[3], lnB[3], ny2, ny2b);
  // ffn + final LN -> d_out
  ffn(nx2b, 0, atted);
  k_add_ln<<<NTOK, 256, 0, stream>>>(nx2, atted, lnG[4], lnB[4], outx, nullptr);
  ffn(ny2b, 1, atted);
  k_add_ln<<<NTOK, 256, 0, stream>>>(ny2, atted, lnG[5], lnB[5], outy, nullptr);
}
